// SparseMoELayer_88811333746987
// MI455X (gfx1250) — matrix operand order repaired and checked by reading, not compiled, not run on hardware
//
#include <hip/hip_runtime.h>
#include <math.h>
#include <stdint.h>

// ---------------- problem constants ----------------
#define TKN   8192      // B*S tokens
#define HID   512       // hidden size
#define DFF_  2048      // expert inner dim
#define NEXP  8         // experts
#define TOPK  2

// ---------------- tiling for expert kernel ----------------
#define MT    128       // token rows per tile
#define CC    32        // DFF chunk (double-buffered)
#define NCH   (DFF_ / CC)
#define TPB   512       // 16 waves
#define XP    520       // sX pitch (elems), 16B-aligned rows, bank-stagger
#define W1P   40        // sW1 pitch (CC=32 + 8 pad)
#define HP    40        // sHdn pitch
#define W2P   520       // sW2 pitch

typedef __attribute__((ext_vector_type(16))) __bf16 v16bf;
typedef __attribute__((ext_vector_type(8)))  float  v8f;
typedef int v4i __attribute__((vector_size(4 * sizeof(int))));

#define AS1 __attribute__((address_space(1)))
#define AS3 __attribute__((address_space(3)))

#if __has_builtin(__builtin_amdgcn_global_load_async_to_lds_b128)
#define HAVE_ASYNC 1
#else
#define HAVE_ASYNC 0
#warning "gfx1250 async-to-LDS builtin NOT available; using synchronous fallback"
#endif

// 16B global -> LDS copy. Async path uses the CDNA5 ASYNCcnt DMA.
__device__ __forceinline__ void cp16(void* lds, const void* g) {
#if HAVE_ASYNC
  __builtin_amdgcn_global_load_async_to_lds_b128(
      (AS1 v4i*)(uintptr_t)g,
      (AS3 v4i*)(uint32_t)(uintptr_t)lds,
      0, 0);
#else
  *(uint4*)lds = *(const uint4*)g;
#endif
}
__device__ __forceinline__ void wait_async0() {
#if HAVE_ASYNC
#if __has_builtin(__builtin_amdgcn_s_wait_asynccnt)
  __builtin_amdgcn_s_wait_asynccnt(0);
#else
  asm volatile("s_wait_asynccnt 0x0" ::: "memory");
#endif
#endif
}

// A-matrix fragment: per-lane 8 bf16 at k0 and 8 bf16 at k0+16 (ISA 16-bit A layout)
__device__ __forceinline__ v16bf ld_fragA(const __bf16* p) {
  union { v16bf v; uint4 u[2]; } r;
  r.u[0] = *(const uint4*)(p);
  r.u[1] = *(const uint4*)(p + 16);
  return r.v;
}
// B-matrix fragment: per-lane column n, 8 k-rows at k0 and 8 at k0+16 (matches A layout)
__device__ __forceinline__ v16bf ld_fragB(const __bf16* p, int pitch) {
  v16bf r;
#pragma unroll
  for (int s = 0; s < 8; ++s) {
    r[s] = p[s * pitch];
    r[s + 8] = p[(s + 16) * pitch];
  }
  return r;
}
__device__ __forceinline__ v8f wmma_bf16(v16bf a, v16bf b, v8f c) {
  return __builtin_amdgcn_wmma_f32_16x16x32_bf16(false, a, false, b, (short)0, c,
                                                 false, false);
}
__device__ __forceinline__ float gelu_tanh(float x) {
  float x3 = x * x * x;
  return 0.5f * x * (1.0f + tanhf(0.7978845608028654f * (x + 0.044715f * x3)));
}

// ---------------- 1) router: logits, top-2 softmax, counts ----------------
__global__ void router_kernel(const float* __restrict__ x,
                              const float* __restrict__ gw,
                              int* __restrict__ topk_idx,
                              float* __restrict__ topk_w,
                              int* __restrict__ counts) {
  const int wave = threadIdx.x >> 5;
  const int lane = threadIdx.x & 31;
  const int t = blockIdx.x * 8 + wave;   // grid 1024 * (8 waves) = 8192 tokens
  if (t >= TKN) return;

  float acc[NEXP];
#pragma unroll
  for (int e = 0; e < NEXP; ++e) acc[e] = 0.0f;
  const float* xr = x + (size_t)t * HID;
  for (int h = lane; h < HID; h += 32) {
    float xv = xr[h];
#pragma unroll
    for (int e = 0; e < NEXP; ++e) acc[e] += xv * gw[e * HID + h];
  }
#pragma unroll
  for (int e = 0; e < NEXP; ++e)
#pragma unroll
    for (int off = 16; off > 0; off >>= 1)
      acc[e] += __shfl_xor(acc[e], off, 32);

  if (lane == 0) {
    float v0 = -3.4e38f, v1 = -3.4e38f;
    int i0 = 0, i1 = 0;
#pragma unroll
    for (int e = 0; e < NEXP; ++e) {
      float v = acc[e];
      if (v > v0) { v1 = v0; i1 = i0; v0 = v; i0 = e; }
      else if (v > v1) { v1 = v; i1 = e; }
    }
    float w0 = 1.0f / (1.0f + expf(v1 - v0));  // softmax over the 2 picks
    topk_idx[t * 2 + 0] = i0;
    topk_idx[t * 2 + 1] = i1;
    topk_w[t * 2 + 0] = w0;
    topk_w[t * 2 + 1] = 1.0f - w0;
    atomicAdd(counts + i0, 1);
    atomicAdd(counts + i1, 1);
  }
}

// ---------------- 2) prefix sum over 8 expert counts ----------------
__global__ void offsets_kernel(const int* __restrict__ counts,
                               int* __restrict__ offsets) {
  if (threadIdx.x == 0) {
    int s = 0;
    for (int e = 0; e < NEXP; ++e) { offsets[e] = s; s += counts[e]; }
    offsets[NEXP] = s;
  }
}

// ---------------- 3) scatter tokens into per-expert segments ----------------
__global__ void scatter_kernel(const int* __restrict__ topk_idx,
                               const float* __restrict__ topk_w,
                               const int* __restrict__ offsets,
                               int* __restrict__ cursors,
                               int* __restrict__ row_ids,
                               float* __restrict__ row_w) {
  int t = blockIdx.x * blockDim.x + threadIdx.x;
  if (t >= TKN) return;
#pragma unroll
  for (int k = 0; k < TOPK; ++k) {
    int e = topk_idx[t * 2 + k];
    int p = atomicAdd(cursors + e, 1);
    int g = offsets[e] + p;
    row_ids[g] = t;
    row_w[g] = topk_w[t * 2 + k];
  }
}

// ---------------- 4) fp32 -> bf16 weight conversion ----------------
__global__ void convert_kernel(const float* __restrict__ w1,
                               const float* __restrict__ w2,
                               __bf16* __restrict__ w1b,
                               __bf16* __restrict__ w2b) {
  const size_t N = (size_t)NEXP * HID * DFF_;
  size_t stride = (size_t)gridDim.x * blockDim.x;
  for (size_t i = (size_t)blockIdx.x * blockDim.x + threadIdx.x; i < N; i += stride) {
    w1b[i] = (__bf16)w1[i];
    w2b[i] = (__bf16)w2[i];
  }
}

// ---------------- 5) fused expert FFN over gathered tokens ----------------
__global__ __launch_bounds__(TPB) void moe_expert_gemm(
    const float* __restrict__ hidden,    // (T,H) fp32
    const float* __restrict__ b1,        // (E,DFF)
    const float* __restrict__ b2,        // (E,H)
    const __bf16* __restrict__ w1b,      // (E,H,DFF)
    const __bf16* __restrict__ w2b,      // (E,DFF,H)
    const int* __restrict__ offsets,     // E+1
    const int* __restrict__ row_ids,     // gathered token ids
    const float* __restrict__ row_w,     // gathered routing weights
    float* __restrict__ out)             // (T,H) fp32, pre-zeroed
{
  __shared__ __bf16 sX[MT * XP];              // gathered X tile, bf16 (133 KB)
  __shared__ __bf16 sW1[2 * HID * W1P];       // W1 chunk dbl-buf: H x CC  (82 KB)
  __shared__ __bf16 sHdn[MT * HP];            // gelu(X@W1)*w, bf16: MT x CC (10 KB)
  __shared__ __bf16 sW2[2 * CC * W2P];        // W2 chunk dbl-buf: CC x H  (67 KB)
  __shared__ float  sRW[MT];
  __shared__ int    sTok[MT];

  const int e = blockIdx.y;
  const int seg_lo = offsets[e], seg_hi = offsets[e + 1];
  const int tile0 = seg_lo + blockIdx.x * MT;
  if (tile0 >= seg_hi) return;       // uniform across block

  const int tid = threadIdx.x;
  const int lane = tid & 31;
  const int wave = tid >> 5;

  const __bf16* w1e = w1b + (size_t)e * HID * DFF_;
  const __bf16* w2e = w2b + (size_t)e * DFF_ * HID;
  const float* b1e = b1 + e * DFF_;

  // async DMA of one (W1chunk, W2chunk) pair into buffer `buf`
  auto load_chunk = [&](int ch, int buf) {
    __bf16* dW1 = sW1 + buf * (HID * W1P);
    __bf16* dW2 = sW2 + buf * (CC * W2P);
#pragma unroll
    for (int it = 0; it < 4; ++it) {
      int idx = tid + it * TPB;              // 2048 x 16B  (512 rows x 4)
      int k = idx >> 2, c = idx & 3;
      cp16(dW1 + k * W1P + c * 8, w1e + (size_t)k * DFF_ + ch * CC + c * 8);
    }
#pragma unroll
    for (int it = 0; it < 4; ++it) {
      int idx = tid + it * TPB;              // 2048 x 16B  (32 rows x 64)
      int k = idx >> 6, c = idx & 63;
      cp16(dW2 + k * W2P + c * 8, w2e + (size_t)(ch * CC + k) * HID + c * 8);
    }
  };

  // kick off chunk 0 DMA before doing the VALU-heavy X staging
  load_chunk(0, 0);

  if (tid < MT) {
    int slot = tile0 + tid;
    int valid = slot < seg_hi;
    sTok[tid] = valid ? row_ids[slot] : -1;
    sRW[tid]  = valid ? row_w[slot] : 0.0f;
  }
  __syncthreads();   // sTok visible for the X staging below

  // stage gathered X rows to LDS as bf16 (4 threads per row, 128 cols each)
  {
    int row = tid >> 2;
    int q = tid & 3;
    int tok = sTok[row];
    const float4* src = (const float4*)(hidden + (size_t)(tok < 0 ? 0 : tok) * HID + q * 128);
    __bf16* dst = sX + row * XP + q * 128;
#pragma unroll 8
    for (int c = 0; c < 32; ++c) {
      float4 v = (tok >= 0) ? src[c] : float4{0.f, 0.f, 0.f, 0.f};
      dst[c * 4 + 0] = (__bf16)v.x;
      dst[c * 4 + 1] = (__bf16)v.y;
      dst[c * 4 + 2] = (__bf16)v.z;
      dst[c * 4 + 3] = (__bf16)v.w;
    }
  }

  // GEMM2 ownership: 4x4 wave grid over 128x512 output (32-row x 128-col blocks)
  const int mb = (wave & 3) * 32;
  const int nb = (wave >> 2) * 128;
  v8f accO[2][8];
#pragma unroll
  for (int i = 0; i < 2; ++i)
#pragma unroll
    for (int j = 0; j < 8; ++j) accO[i][j] = v8f{0, 0, 0, 0, 0, 0, 0, 0};

  // GEMM1 ownership: 8 m-tiles x 2 n-tiles over 128 x CC -> one 16x16 tile per wave
  const int mt1 = wave & 7;
  const int nt1 = wave >> 3;
  const int ml = lane & 15;
  const int kb = (lane >> 4) * 8;
  const int mhi = 8 * (lane >> 4);

  for (int ch = 0; ch < NCH; ++ch) {
    const int buf = ch & 1;
    wait_async0();
    __syncthreads();   // chunk `ch` in LDS; previous readers of sHdn/other buf done

    if (ch + 1 < NCH) load_chunk(ch + 1, buf ^ 1);   // overlap DMA with WMMA

    // ---- GEMM1: Hdn = gelu(X @ W1chunk + b1) * row_weight ----
    {
      v8f a0 = v8f{0, 0, 0, 0, 0, 0, 0, 0};
      const __bf16* arow = sX + (mt1 * 16 + ml) * XP + kb;
      const __bf16* bbase = sW1 + buf * (HID * W1P) + nt1 * 16;
#pragma unroll
      for (int ks = 0; ks < HID / 32; ++ks) {
        v16bf af = ld_fragA(arow + ks * 32);
        v16bf bf0 = ld_fragB(bbase + (ks * 32 + kb) * W1P + ml, W1P);
        a0 = wmma_bf16(af, bf0, a0);
      }
      float bb = b1e[ch * CC + nt1 * 16 + ml];
#pragma unroll
      for (int r = 0; r < 8; ++r) {
        int m = mt1 * 16 + mhi + r;
        sHdn[m * HP + nt1 * 16 + ml] = (__bf16)(gelu_tanh(a0[r] + bb) * sRW[m]);
      }
    }
    __syncthreads();   // sHdn ready

    // ---- GEMM2: accO += Hdn(mb.., :CC) @ W2chunk(:CC, nb..) ----  (one k-step)
    {
      v16bf aF0 = ld_fragA(sHdn + (mb + ml) * HP + kb);
      v16bf aF1 = ld_fragA(sHdn + (mb + 16 + ml) * HP + kb);
      const __bf16* brow = sW2 + buf * (CC * W2P) + kb * W2P + nb + ml;
#pragma unroll
      for (int j = 0; j < 8; ++j) {
        v16bf bF = ld_fragB(brow + j * 16, W2P);
        accO[0][j] = wmma_bf16(aF0, bF, accO[0][j]);
        accO[1][j] = wmma_bf16(aF1, bF, accO[1][j]);
      }
    }
  }

  // epilogue: out[token, n] += accO + w * b2[n]   (2 commutative f32 atomics/elem)
  {
    const float* b2e = b2 + e * HID;
#pragma unroll
    for (int i = 0; i < 2; ++i) {
#pragma unroll
      for (int r = 0; r < 8; ++r) {
        int m = mb + i * 16 + mhi + r;
        int tok = sTok[m];
        if (tok < 0) continue;
        float w = sRW[m];
        float* orow = out + (size_t)tok * HID;
#pragma unroll
        for (int j = 0; j < 8; ++j) {
          int n = nb + j * 16 + ml;
          atomicAdd(orow + n, accO[i][j][r] + w * b2e[n]);
        }
      }
    }
  }
}

// ---------------- host launch ----------------
extern "C" void kernel_launch(void* const* d_in, const int* in_sizes, int n_in,
                              void* d_out, int out_size, void* d_ws, size_t ws_size,
                              hipStream_t stream) {
  const float* hidden = (const float*)d_in[0];  // (B,S,H)
  const float* gate_w = (const float*)d_in[1];  // (E,H)
  const float* w1     = (const float*)d_in[2];  // (E,H,DFF)
  const float* b1     = (const float*)d_in[3];  // (E,DFF)
  const float* w2     = (const float*)d_in[4];  // (E,DFF,H)
  const float* b2     = (const float*)d_in[5];  // (E,H)
  float* out = (float*)d_out;

  // workspace layout (~33.8 MB total)
  char* ws = (char*)d_ws;
  int*    counts   = (int*)(ws + 0);           // 8 ints
  int*    cursors  = (int*)(ws + 32);          // 8 ints
  int*    offsets  = (int*)(ws + 64);          // 9 ints
  int*    topk_idx = (int*)(ws + 256);                      // T*2 ints
  float*  topk_w   = (float*)(ws + 256 + 1 * 65536);        // T*2 floats
  int*    row_ids  = (int*)(ws + 256 + 2 * 65536);          // T*2 ints
  float*  row_w    = (float*)(ws + 256 + 3 * 65536);        // T*2 floats
  __bf16* w1b      = (__bf16*)(ws + 262400);                // 16 MB
  __bf16* w2b      = (__bf16*)(ws + 262400 + 16777216);     // 16 MB

  (void)hipMemsetAsync(ws, 0, 256, stream);                                 // counts/cursors
  (void)hipMemsetAsync(d_out, 0, (size_t)out_size * sizeof(float), stream); // atomic target

  router_kernel<<<TKN / 8, 256, 0, stream>>>(hidden, gate_w, topk_idx, topk_w, counts);
  offsets_kernel<<<1, 32, 0, stream>>>(counts, offsets);
  scatter_kernel<<<TKN / 256, 256, 0, stream>>>(topk_idx, topk_w, offsets, cursors,
                                                row_ids, row_w);
  convert_kernel<<<4096, 256, 0, stream>>>(w1, w2, w1b, w2b);
  // worst case one expert owns all 8192 tokens -> 64 tiles of 128
  moe_expert_gemm<<<dim3(64, NEXP), TPB, 0, stream>>>(hidden, b1, b2, w1b, w2b,
                                                      offsets, row_ids, row_w, out);
}
